// SpatialAttention_73186242724259
// MI455X (gfx1250) — matrix operand order repaired and checked by reading, not compiled, not run on hardware
//
#include <hip/hip_runtime.h>
#include <hip/hip_bf16.h>
#include <cstdint>

typedef __attribute__((ext_vector_type(16))) __bf16       v16bf;
typedef __attribute__((ext_vector_type(8)))  float        v8f;
typedef __attribute__((ext_vector_type(4)))  unsigned int u32x4;
typedef __attribute__((ext_vector_type(4)))  float        f32x4;

#define Bn   8
#define Tn   32
#define Cn   128
#define An   32
#define HWn  4096
#define KP   136   // LDS kv-tile pitch in bf16 elems (17*16B rows -> conflict-free, 16B aligned)
#define SP   132   // LDS output-stage pitch in floats
#define POS_TILE 64
#define THREADS  128

union FragAB { v16bf bf; u32x4 q[2]; };

__device__ __forceinline__ unsigned short f2bf(float f) {
  union { float f; unsigned u; } x; x.f = f;
  unsigned r = x.u + 0x7FFFu + ((x.u >> 16) & 1u);   // round-to-nearest-even
  return (unsigned short)(r >> 16);
}
__device__ __forceinline__ unsigned pack2(float a, float b) {
  return (unsigned)f2bf(a) | ((unsigned)f2bf(b) << 16);
}

// Convert fp32 weights to bf16 scratch: [Wk 32x128 | Wq 32x128 | Wv 128x128]
__global__ void prep_weights_kernel(const float* __restrict__ Wk,
                                    const float* __restrict__ Wq,
                                    const float* __restrict__ Wv,
                                    unsigned short* __restrict__ w) {
  int i = blockIdx.x * blockDim.x + threadIdx.x;
  if (i < 4096)       w[i] = f2bf(Wk[i]);
  else if (i < 8192)  w[i] = f2bf(Wq[i - 4096]);
  else if (i < 24576) w[i] = f2bf(Wv[i - 8192]);
}

__global__ __launch_bounds__(THREADS) void spatial_attn_kernel(
    const float* __restrict__ query,
    const float* __restrict__ kv,
    const float* __restrict__ bq,
    const float* __restrict__ bk,
    const float* __restrict__ Ws,
    const float* __restrict__ bs,
    const float* __restrict__ bv,
    const float* __restrict__ gamma,
    const unsigned short* __restrict__ w,   // bf16 [Wk|Wq|Wv]
    float* __restrict__ out)
{
  extern __shared__ char smem[];
  unsigned short* kvt = (unsigned short*)smem;                              // 64 x KP bf16
  float* stg = (float*)(smem + (size_t)POS_TILE * KP * sizeof(unsigned short)); // 64 x SP f32

  const int tid  = threadIdx.x;
  const int b    = blockIdx.x >> 6;
  const int pos0 = (blockIdx.x & 63) * POS_TILE;

  const int lane = tid & 31;
  const int wave = tid >> 5;
  const int half = lane >> 4;     // which 16-lane half
  const int nn   = lane & 15;     // N column within tile
  const int mrow = lane & 15;     // A-matrix row

  const unsigned short* wk = w;
  const unsigned short* wq = w + 4096;
  const unsigned short* wv = w + 8192;

  const float ws0 = Ws[nn];
  const float ws1 = Ws[16 + nn];
  const float bs0 = bs[0];
  float bvn[8];
#pragma unroll
  for (int nt = 0; nt < 8; ++nt) bvn[nt] = bv[nt * 16 + nn];

  const int sub = tid >> 4;        // 0..7 : c-pair group for staging
  const int px  = (tid & 15) * 4;  // 0..60: position chunk (float4)

  // ---- stage query tile into LDS as bf16 [pos][c] ----
  {
    const float* src = query + (size_t)b * Cn * HWn + pos0;
#pragma unroll
    for (int it = 0; it < 8; ++it) {
      int c = it * 16 + sub * 2;
      f32x4 r0 = *(const f32x4*)(src + (size_t)c * HWn + px);
      f32x4 r1 = *(const f32x4*)(src + (size_t)(c + 1) * HWn + px);
#pragma unroll
      for (int j = 0; j < 4; ++j)
        *(unsigned*)(kvt + (size_t)(px + j) * KP + c) = pack2(r0[j], r1[j]);
    }
  }
  __syncthreads();

  unsigned short* myKv = kvt + (size_t)(wave * 16) * KP;

  // ---- Qp = Xq * Wq^T, fold in (bq + bk) ----
  v8f qp[2];
#pragma unroll
  for (int nt = 0; nt < 2; ++nt) {
    v8f d = {};
#pragma unroll
    for (int kt = 0; kt < 4; ++kt) {
      FragAB a, bw;
      a.q[0] = *(const u32x4*)(myKv + mrow * KP + kt * 32 + half * 8);
      a.q[1] = *(const u32x4*)(myKv + mrow * KP + kt * 32 + 16 + half * 8);
      const u32x4* p = (const u32x4*)(wq + (size_t)(nt * 16 + nn) * 128 + kt * 32 + half * 8);
      bw.q[0] = p[0]; bw.q[1] = p[2];
      d = __builtin_amdgcn_wmma_f32_16x16x32_bf16(false, a.bf, false, bw.bf, (short)0, d, false, false);
    }
    float bb = bq[nt * 16 + nn] + bk[nt * 16 + nn];
#pragma unroll
    for (int v = 0; v < 8; ++v) d[v] += bb;
    qp[nt] = d;
  }

  // online-softmax state: rows r = half*8 + v
  float mr[8], lr[8];
  v8f acc[8];
#pragma unroll
  for (int v = 0; v < 8; ++v) { mr[v] = -3.0e38f; lr[v] = 0.f; }
#pragma unroll
  for (int nt = 0; nt < 8; ++nt)
#pragma unroll
    for (int v = 0; v < 8; ++v) acc[nt][v] = 0.f;

  for (int t = 0; t < Tn; ++t) {
    __syncthreads();
    // ---- cooperative coalesced load of kv[b,t] tile -> LDS bf16 [pos][c] ----
    {
      const float* src  = kv + ((size_t)(b * Tn + t) * Cn) * HWn + pos0;
      const float* nsrc = (t + 1 < Tn) ? src + (size_t)Cn * HWn : nullptr;
#pragma unroll
      for (int it = 0; it < 8; ++it) {
        int c = it * 16 + sub * 2;
        f32x4 r0 = *(const f32x4*)(src + (size_t)c * HWn + px);
        f32x4 r1 = *(const f32x4*)(src + (size_t)(c + 1) * HWn + px);
        if (nsrc) __builtin_prefetch(nsrc + (size_t)c * HWn + px, 0, 0);
#pragma unroll
        for (int j = 0; j < 4; ++j)
          *(unsigned*)(kvt + (size_t)(px + j) * KP + c) = pack2(r0[j], r1[j]);
      }
    }
    __syncthreads();

    // A-fragments for this wave's 16 positions (documented 16-bit A 16x32 layout)
    FragAB af[4];
#pragma unroll
    for (int kt = 0; kt < 4; ++kt) {
      af[kt].q[0] = *(const u32x4*)(myKv + mrow * KP + kt * 32 + half * 8);
      af[kt].q[1] = *(const u32x4*)(myKv + mrow * KP + kt * 32 + 16 + half * 8);
    }

    // ---- Kp, then score contribution sum_a ws[a]*tanh(Qp+Kp) ----
    float sc[8];
#pragma unroll
    for (int v = 0; v < 8; ++v) sc[v] = 0.f;
#pragma unroll
    for (int nt = 0; nt < 2; ++nt) {
      v8f d = {};
#pragma unroll
      for (int kt = 0; kt < 4; ++kt) {
        FragAB bw;
        const u32x4* p = (const u32x4*)(wk + (size_t)(nt * 16 + nn) * 128 + kt * 32 + half * 8);
        bw.q[0] = p[0]; bw.q[1] = p[2];
        d = __builtin_amdgcn_wmma_f32_16x16x32_bf16(false, af[kt].bf, false, bw.bf, (short)0, d, false, false);
      }
      float wsn = nt ? ws1 : ws0;
#pragma unroll
      for (int v = 0; v < 8; ++v) sc[v] += wsn * tanhf(qp[nt][v] + d[v]);
    }
    // reduce over the 16 N-lanes (a-dimension lives within each half)
#pragma unroll
    for (int off = 1; off < 16; off <<= 1)
#pragma unroll
      for (int v = 0; v < 8; ++v) sc[v] += __shfl_xor(sc[v], off, 16);

    // ---- online softmax update ----
    float pv[8], al[8];
#pragma unroll
    for (int v = 0; v < 8; ++v) {
      float s  = sc[v] + bs0;
      float mn = fmaxf(mr[v], s);
      al[v] = __expf(mr[v] - mn);
      pv[v] = __expf(s - mn);
      lr[v] = lr[v] * al[v] + pv[v];
      mr[v] = mn;
    }

    // ---- Vp (16x128) and fused accumulate ----
#pragma unroll
    for (int nt = 0; nt < 8; ++nt) {
      v8f d = {};
#pragma unroll
      for (int kt = 0; kt < 4; ++kt) {
        FragAB bw;
        const u32x4* p = (const u32x4*)(wv + (size_t)(nt * 16 + nn) * 128 + kt * 32 + half * 8);
        bw.q[0] = p[0]; bw.q[1] = p[2];
        d = __builtin_amdgcn_wmma_f32_16x16x32_bf16(false, af[kt].bf, false, bw.bf, (short)0, d, false, false);
      }
      float bvv = bvn[nt];
#pragma unroll
      for (int v = 0; v < 8; ++v)
        acc[nt][v] = acc[nt][v] * al[v] + pv[v] * (d[v] + bvv);
    }
  }

  // ---- finalize: ctx = acc / l, stage to LDS [pos][c] ----
  float inv[8];
#pragma unroll
  for (int v = 0; v < 8; ++v) inv[v] = 1.0f / lr[v];
#pragma unroll
  for (int nt = 0; nt < 8; ++nt)
#pragma unroll
    for (int v = 0; v < 8; ++v)
      stg[(size_t)(wave * 16 + half * 8 + v) * SP + nt * 16 + nn] = acc[nt][v] * inv[v];
  __syncthreads();

  // ---- coalesced epilogue: out = gamma*ctx + query ----
  const float g = gamma[0];
  const float* qb = query + (size_t)b * Cn * HWn + pos0;
  float* ob = out + (size_t)b * Cn * HWn + pos0;
#pragma unroll
  for (int it = 0; it < 16; ++it) {
    int c = it * 8 + sub;
    f32x4 qv = *(const f32x4*)(qb + (size_t)c * HWn + px);
    f32x4 o;
#pragma unroll
    for (int j = 0; j < 4; ++j)
      o[j] = g * stg[(size_t)(px + j) * SP + c] + qv[j];
    *(f32x4*)(ob + (size_t)c * HWn + px) = o;
  }
}

extern "C" void kernel_launch(void* const* d_in, const int* in_sizes, int n_in,
                              void* d_out, int out_size, void* d_ws, size_t ws_size,
                              hipStream_t stream) {
  const float* query = (const float*)d_in[0];
  const float* kvs   = (const float*)d_in[1];
  const float* Wq    = (const float*)d_in[2];
  const float* bq    = (const float*)d_in[3];
  const float* Wk    = (const float*)d_in[4];
  const float* bk    = (const float*)d_in[5];
  const float* Ws    = (const float*)d_in[6];
  const float* bs    = (const float*)d_in[7];
  const float* Wv    = (const float*)d_in[8];
  const float* bv    = (const float*)d_in[9];
  const float* gamma = (const float*)d_in[10];
  float* out = (float*)d_out;
  unsigned short* w = (unsigned short*)d_ws;   // 24576 bf16 = 48KB scratch

  prep_weights_kernel<<<(24576 + 255) / 256, 256, 0, stream>>>(Wk, Wq, Wv, w);

  size_t shmem = (size_t)POS_TILE * KP * sizeof(unsigned short)
               + (size_t)POS_TILE * SP * sizeof(float);   // 51,200 B
  dim3 grid(Bn * (HWn / POS_TILE));                       // 512 workgroups
  spatial_attn_kernel<<<grid, THREADS, shmem, stream>>>(query, kvs, bq, bk, Ws, bs, bv, gamma, w, out);
}
